// LSTMModel_7172595384877
// MI455X (gfx1250) — matrix operand order repaired and checked by reading, not compiled, not run on hardware
//
#include <hip/hip_runtime.h>

typedef __attribute__((ext_vector_type(16))) _Float16 v16h;
typedef __attribute__((ext_vector_type(8)))  _Float16 v8h;
typedef __attribute__((ext_vector_type(2)))  _Float16 v2h;
typedef __attribute__((ext_vector_type(8)))  float    v8f;

#define B_SZ 4096
#define T_SZ 512
#define D_SZ 8
#define H_SZ 64
#define G_SZ 256   // 4*H
#define BM   16    // batch rows per block
#define NTHR 64    // 2 waves: each owns 2 of the 4 k-quads (all 4 gates)

static __device__ __forceinline__ float tanh_fallback(float x) {
  float e = __expf(-2.0f * fabsf(x));
  float r = (1.0f - e) / (1.0f + e);
  return copysignf(r, x);
}

#if defined(__has_builtin)
#if __has_builtin(__builtin_amdgcn_tanhf)
#define DEV_TANH(x) __builtin_amdgcn_tanhf(x)
#define HAVE_HW_TANH 1
#endif
#endif
#ifndef DEV_TANH
#define DEV_TANH(x) tanh_fallback(x)
#endif

static __device__ __forceinline__ float sigm(float x) {
#ifdef HAVE_HW_TANH
  return __builtin_fmaf(0.5f, DEV_TANH(0.5f * x), 0.5f);
#else
  return 1.0f / (1.0f + __expf(-x));
#endif
}

// A-matrix fragment (16x32 f16, ISA 7.12.2): lane m = lane%16, hs = lane/16.
// elements 0..7  <- K = kc + hs*8 .. +7      (one ds_load_b128)
// elements 8..15 <- K = kc + 16 + hs*8 .. +7 (one ds_load_b128)
static __device__ __forceinline__ v16h load_a_frag(const _Float16* row, int kc, int hs) {
  v8h lo = *(const v8h*)(row + kc + hs * 8);
  v8h hi = *(const v8h*)(row + kc + 16 + hs * 8);
  return __builtin_shufflevector(lo, hi, 0,1,2,3,4,5,6,7,8,9,10,11,12,13,14,15);
}

// B-matrix fragment (32x16 f16): lane n = lane%16, element e -> K = hs*8 + e (e<8), 16 + hs*8 + (e-8) (e>=8).
// Loaded once from global fp32 (row-major, this tile's row = W + n*stride + kc).
static __device__ __forceinline__ v16h load_bfrag_g(const float* Wrow, int hs) {
  const float* p = Wrow + hs * 8;
  v16h f;
  #pragma unroll
  for (int e = 0; e < 16; ++e) f[e] = (_Float16)p[e < 8 ? e : e + 8];
  return f;
}

static __device__ __forceinline__ v8f wmma_f16(v16h a, v16h b, v8f c) {
  return __builtin_amdgcn_wmma_f32_16x16x32_f16(false, a, false, b, (short)0, c, false, false);
}

__global__ void __launch_bounds__(NTHR, 1) lstm2_persist(
    const float* __restrict__ x,
    const float* __restrict__ Wih0, const float* __restrict__ Whh0,
    const float* __restrict__ bih0, const float* __restrict__ bhh0,
    const float* __restrict__ Wih1, const float* __restrict__ Whh1,
    const float* __restrict__ bih1, const float* __restrict__ bhh1,
    const float* __restrict__ Wfc,  const float* __restrict__ bfc,
    float* __restrict__ out)
{
  __shared__ _Float16 sX[BM * 32];      // x_t, K zero-padded 8->32
  __shared__ _Float16 sH0[BM * H_SZ];   // layer0 hidden (cross-wave K exchange)
  __shared__ _Float16 sH1[BM * H_SZ];   // layer1 hidden

  const int tid  = threadIdx.x;
  const int wid  = tid >> 5;    // 0,1 : owns k-quads {2*wid, 2*wid+1}
  const int lane = tid & 31;
  const int ln   = lane & 15;
  const int hs   = lane >> 4;
  const int q0   = wid * 2;

  for (int i = tid; i < BM * 32; i += NTHR) sX[i] = (_Float16)0.0f;
  for (int i = tid; i < BM * H_SZ; i += NTHR) {
    sH0[i] = (_Float16)0.0f;
    sH1[i] = (_Float16)0.0f;
  }

  // ---- persistent weight fragments: global fp32 -> fp16 VGPRs, once ----
  v16h bIh0[2][4];       // Wih0, K=0..31 (cols >= 8 zero)
  v16h bHh0[2][4][2];    // Whh0, K chunks 0..31 / 32..63
  v16h bIh1[2][4][2];    // Wih1
  v16h bHh1[2][4][2];    // Whh1
  v8f  bspl0[2][4], bspl1[2][4];   // loop-invariant splatted biases (WMMA C seed)

  #pragma unroll
  for (int qi = 0; qi < 2; ++qi) {
    #pragma unroll
    for (int g = 0; g < 4; ++g) {
      const int n = g * H_SZ + (q0 + qi) * 16 + ln;   // gate column owned by this lane
      const float b0 = bih0[n] + bhh0[n];
      const float b1 = bih1[n] + bhh1[n];
      bspl0[qi][g] = (v8f){ b0, b0, b0, b0, b0, b0, b0, b0 };
      bspl1[qi][g] = (v8f){ b1, b1, b1, b1, b1, b1, b1, b1 };
      v16h f;
      #pragma unroll
      for (int e = 0; e < 16; ++e) {
        const int K = (e < 8) ? hs * 8 + e : 16 + hs * 8 + (e - 8);
        f[e] = (K < D_SZ) ? (_Float16)Wih0[n * D_SZ + K] : (_Float16)0.0f;
      }
      bIh0[qi][g] = f;
      #pragma unroll
      for (int kc = 0; kc < 2; ++kc) {
        bHh0[qi][g][kc] = load_bfrag_g(Whh0 + n * H_SZ + kc * 32, hs);
        bIh1[qi][g][kc] = load_bfrag_g(Wih1 + n * H_SZ + kc * 32, hs);
        bHh1[qi][g][kc] = load_bfrag_g(Whh1 + n * H_SZ + kc * 32, hs);
      }
    }
  }
  __syncthreads();

  float c0r[2][8], c1r[2][8];
  #pragma unroll
  for (int qi = 0; qi < 2; ++qi)
    #pragma unroll
    for (int v = 0; v < 8; ++v) { c0r[qi][v] = 0.0f; c1r[qi][v] = 0.0f; }

  // x_t loader: thread covers (row = tid/4, 2-float slice = tid%4)
  const int xrow = tid >> 2, xpart = tid & 3;
  const float* xp = x + ((long)(blockIdx.x * BM + xrow) * T_SZ) * D_SZ + xpart * 2;
  _Float16* xdst = sX + xrow * 32 + xpart * 2;

  #pragma unroll 1
  for (int t = 0; t < T_SZ; ++t) {
    float2 xv = *(const float2*)(xp + (long)t * D_SZ);
    __builtin_prefetch(xp + (long)(t + 1) * D_SZ, 0, 0);
    v2h hx = { (_Float16)xv.x, (_Float16)xv.y };
    *(v2h*)xdst = hx;

    // h0_{t-1} fragments: must be read before anyone stores h0_t (B1 guards)
    v16h ah0a = load_a_frag(sH0 + ln * H_SZ, 0,  hs);
    v16h ah0b = load_a_frag(sH0 + ln * H_SZ, 32, hs);
    __syncthreads();   // B1: x_t + h1_{t-1} stores visible; old h0 consumed

    // x_t and h1_{t-1} fragments: visibility guarded by B1, overwrite by B2
    v16h ax   = load_a_frag(sX  + ln * 32,   0,  hs);
    v16h ah1a = load_a_frag(sH1 + ln * H_SZ, 0,  hs);
    v16h ah1b = load_a_frag(sH1 + ln * H_SZ, 32, hs);

    // ================= layer 0 =================
    #pragma unroll
    for (int qi = 0; qi < 2; ++qi) {
      v8f acc[4];
      #pragma unroll
      for (int g = 0; g < 4; ++g) {
        v8f a = wmma_f16(ax,   bIh0[qi][g],    bspl0[qi][g]);  // C seeded with bias
        a     = wmma_f16(ah0a, bHh0[qi][g][0], a);
        a     = wmma_f16(ah0b, bHh0[qi][g][1], a);
        acc[g] = a;
      }
      #pragma unroll
      for (int v = 0; v < 8; ++v) {
        float ig = sigm(acc[0][v]);
        float fg = sigm(acc[1][v]);
        float gv = DEV_TANH(acc[2][v]);
        float og = sigm(acc[3][v]);
        float c  = fg * c0r[qi][v] + ig * gv;
        c0r[qi][v] = c;
        sH0[(hs * 8 + v) * H_SZ + (q0 + qi) * 16 + ln] = (_Float16)(og * DEV_TANH(c));
      }
    }
    __syncthreads();   // B2: h0_t complete; ax/ah1 consumed before overwrite

    v16h ai0 = load_a_frag(sH0 + ln * H_SZ, 0,  hs);
    v16h ai1 = load_a_frag(sH0 + ln * H_SZ, 32, hs);

    // ================= layer 1 =================
    #pragma unroll
    for (int qi = 0; qi < 2; ++qi) {
      v8f acc[4];
      #pragma unroll
      for (int g = 0; g < 4; ++g) {
        v8f a = wmma_f16(ai0,  bIh1[qi][g][0], bspl1[qi][g]);  // C seeded with bias
        a     = wmma_f16(ai1,  bIh1[qi][g][1], a);
        a     = wmma_f16(ah1a, bHh1[qi][g][0], a);
        a     = wmma_f16(ah1b, bHh1[qi][g][1], a);
        acc[g] = a;
      }
      #pragma unroll
      for (int v = 0; v < 8; ++v) {
        float ig = sigm(acc[0][v]);
        float fg = sigm(acc[1][v]);
        float gv = DEV_TANH(acc[2][v]);
        float og = sigm(acc[3][v]);
        float c  = fg * c1r[qi][v] + ig * gv;
        c1r[qi][v] = c;
        sH1[(hs * 8 + v) * H_SZ + (q0 + qi) * 16 + ln] = (_Float16)(og * DEV_TANH(c));
      }
    }
    // no loop-end barrier needed: next B1 provides h1_t visibility,
    // next B2 guards h1 fragment consumption before overwrite
  }
  __syncthreads();   // h1 last step visible for the FC readout

  // ---- final FC: out[m] = h1_last[m] . Wfc + bfc ----
  if (tid < BM) {
    float s = bfc[0];
    const _Float16* hrow = sH1 + tid * H_SZ;
    #pragma unroll 8
    for (int k = 0; k < H_SZ; ++k) s += (float)hrow[k] * Wfc[k];
    out[(long)blockIdx.x * BM + tid] = s;
  }
}

extern "C" void kernel_launch(void* const* d_in, const int* in_sizes, int n_in,
                              void* d_out, int out_size, void* d_ws, size_t ws_size,
                              hipStream_t stream) {
  (void)in_sizes; (void)n_in; (void)out_size; (void)d_ws; (void)ws_size;
  const float* x    = (const float*)d_in[0];
  const float* Wih0 = (const float*)d_in[1];
  const float* Whh0 = (const float*)d_in[2];
  const float* bih0 = (const float*)d_in[3];
  const float* bhh0 = (const float*)d_in[4];
  const float* Wih1 = (const float*)d_in[5];
  const float* Whh1 = (const float*)d_in[6];
  const float* bih1 = (const float*)d_in[7];
  const float* bhh1 = (const float*)d_in[8];
  const float* Wfc  = (const float*)d_in[9];
  const float* bfc  = (const float*)d_in[10];
  float* out = (float*)d_out;

  lstm2_persist<<<dim3(B_SZ / BM), dim3(NTHR), 0, stream>>>(
      x, Wih0, Whh0, bih0, bhh0, Wih1, Whh1, bih1, bhh1, Wfc, bfc, out);
}
